// RDN_71786083385597
// MI455X (gfx1250) — matrix operand order repaired and checked by reading, not compiled, not run on hardware
//
#include <hip/hip_runtime.h>
#include <hip/hip_bf16.h>

typedef __attribute__((ext_vector_type(16))) __bf16 v16bf;
typedef __attribute__((ext_vector_type(8)))  float  v8f;

#define TEMP 34.0f
#define SLOPE 0.2f
#define Bn 16
#define Cn 64
#define Hn 256
#define Wn 256
#define Kn 4
#define On 64
#define HID 17
#define TW 64          // pixel tile width
#define SW 66          // staged width (with halo)
#define STAGE_ELEMS (3*SW*Cn)   // 12672

static __device__ __forceinline__ unsigned short f32_to_bf16(float f) {
    unsigned int u = __float_as_uint(f);
    u += 0x7FFFu + ((u >> 16) & 1u);          // round to nearest even
    return (unsigned short)(u >> 16);
}
static __device__ __forceinline__ float bf16_to_f32(unsigned short h) {
    return __uint_as_float(((unsigned int)h) << 16);
}

// ---------------- 1) global average pool: pooled[b*C+c] ----------------
__global__ __launch_bounds__(256) void pool_kernel(const float* __restrict__ x,
                                                   float* __restrict__ pooled) {
    __shared__ float red[256];
    const int bc = blockIdx.x;                 // 0..B*C-1
    const float* p = x + (size_t)bc * (Hn * Wn);
    float s = 0.f;
    for (int i = threadIdx.x; i < Hn * Wn; i += 256) s += p[i];
    red[threadIdx.x] = s;
    __syncthreads();
    for (int off = 128; off > 0; off >>= 1) {
        if (threadIdx.x < off) red[threadIdx.x] += red[threadIdx.x + off];
        __syncthreads();
    }
    if (threadIdx.x == 0) pooled[bc] = red[0] * (1.0f / (Hn * Wn));
}

// ---------------- 2) attention + aggregated bias ----------------
__global__ void attn_kernel(const float* __restrict__ pooled,
                            const float* __restrict__ fc1_w,
                            const float* __restrict__ fc2_w,
                            const float* __restrict__ fc2_b,
                            const float* __restrict__ bias_k,
                            float* __restrict__ att,
                            float* __restrict__ agg_b) {
    const int b = threadIdx.x;
    if (b >= Bn) return;
    float h[HID];
    for (int j = 0; j < HID; ++j) {
        float s = 0.f;
        for (int c = 0; c < Cn; ++c) s += pooled[b * Cn + c] * fc1_w[j * Cn + c];
        h[j] = (s >= 0.f) ? s : SLOPE * s;
    }
    float lg[Kn], m = -1e30f;
    for (int k = 0; k < Kn; ++k) {
        float s = fc2_b[k];
        for (int j = 0; j < HID; ++j) s += h[j] * fc2_w[k * HID + j];
        s *= (1.0f / TEMP);
        lg[k] = s;
        m = fmaxf(m, s);
    }
    float e[Kn], den = 0.f;
    for (int k = 0; k < Kn; ++k) { e[k] = __expf(lg[k] - m); den += e[k]; }
    for (int k = 0; k < Kn; ++k) { e[k] /= den; att[b * Kn + k] = e[k]; }
    for (int o = 0; o < On; ++o) {
        float s = 0.f;
        for (int k = 0; k < Kn; ++k) s += e[k] * bias_k[k * On + o];
        agg_b[b * On + o] = s;
    }
}

// ------- 3) aggregated weights, prepacked in WMMA A-fragment lane order -------
// layout: idx = ((((b*9+kp)*2+ch)*4+otile)*32+lane)*16 + j   (bf16 hi/lo planes)
__global__ __launch_bounds__(256) void afrag_kernel(const float* __restrict__ weight,
                                                    const float* __restrict__ att,
                                                    unsigned short* __restrict__ Ahi,
                                                    unsigned short* __restrict__ Alo) {
    const int idx = blockIdx.x * 256 + threadIdx.x;      // < 16*9*2*4*32*16 = 589824
    const int j     = idx & 15;
    int t = idx >> 4;
    const int lane  = t & 31;  t >>= 5;
    const int otile = t & 3;   t >>= 2;
    const int ch    = t & 1;   t >>= 1;
    const int kp    = t % 9;
    const int b     = t / 9;
    // A-matrix 16x32 bf16 slot->K map: contiguous, k = 16*(lane>=16) + j (matches B)
    const int klocal = j + ((lane >= 16) ? 16 : 0);
    const int i = ch * 32 + klocal;
    const int o = otile * 16 + (lane & 15);
    float s = 0.f;
    for (int k = 0; k < Kn; ++k)
        s += att[b * Kn + k] * weight[(((size_t)k * On + o) * Cn + i) * 9 + kp];
    const unsigned short hi = f32_to_bf16(s);
    const unsigned short lo = f32_to_bf16(s - bf16_to_f32(hi));
    Ahi[idx] = hi;
    Alo[idx] = lo;
}

// ---------------- 4) implicit-GEMM conv via bf16x3 WMMA ----------------
__global__ __launch_bounds__(256) void conv_kernel(const float* __restrict__ x,
                                                   const unsigned short* __restrict__ Ahi,
                                                   const unsigned short* __restrict__ Alo,
                                                   const float* __restrict__ agg_b,
                                                   float* __restrict__ out) {
    __shared__ __align__(32) unsigned short sh_hi[STAGE_ELEMS];
    __shared__ __align__(32) unsigned short sh_lo[STAGE_ELEMS];

    const int tid = threadIdx.x;
    const int b   = blockIdx.y;
    const int ty  = blockIdx.x >> 2;          // output row 0..255
    const int x0  = (blockIdx.x & 3) << 6;    // pixel tile base
    const float* xb = x + (size_t)b * Cn * Hn * Wn;

    // stage input rows ty-1..ty+1, cols x0-1..x0+64, all 64 ch, split hi/lo bf16
    for (int e = tid; e < STAGE_ELEMS; e += 256) {
        const int c = e % SW;
        const int t2 = e / SW;
        const int r = t2 % 3;
        const int i = t2 / 3;
        const int gy = ty + r - 1;
        const int gx = x0 + c - 1;
        float v = 0.f;
        if ((unsigned)gy < (unsigned)Hn && (unsigned)gx < (unsigned)Wn)
            v = xb[((size_t)i << 16) + (gy << 8) + gx];
        const unsigned short hi = f32_to_bf16(v);
        const unsigned short lo = f32_to_bf16(v - bf16_to_f32(hi));
        const int li = (r * SW + c) * Cn + i;   // channel-contiguous per (row,col)
        sh_hi[li] = hi;
        sh_lo[li] = lo;
    }
    __syncthreads();

    const int lane  = tid & 31;
    const int wave  = tid >> 5;
    const int otile = wave & 3;               // 0..3 -> output channels otile*16..+15
    const int pxh   = wave >> 2;              // 0..1 -> pixel half of the 64-wide tile
    const int n16   = lane & 15;              // fragment column
    const int koff  = lane & 16;              // B-fragment: lanes>=16 hold K 16..31

    v8f acc0 = {};                            // pixels pxh*32 + [0..15]
    v8f acc1 = {};                            // pixels pxh*32 + [16..31]

    for (int kp = 0; kp < 9; ++kp) {
        const int kh = kp / 3;
        const int kw = kp - kh * 3;
#pragma unroll
        for (int ch = 0; ch < 2; ++ch) {
            const size_t offA =
                ((((((size_t)b * 9 + kp) * 2 + ch) * 4 + otile) * 32) + lane) * 16;
            const v16bf a_h = *(const v16bf*)(Ahi + offA);
            const v16bf a_l = *(const v16bf*)(Alo + offA);

            const int kb = ch * 32 + koff;    // 16 contiguous input channels
            const int n0 = (pxh << 5) + n16;
            const int base0 = (kh * SW + (n0 + kw)) * Cn + kb;
            const int base1 = base0 + 16 * Cn;

            const v16bf b0h = *(const v16bf*)(sh_hi + base0);
            const v16bf b0l = *(const v16bf*)(sh_lo + base0);
            acc0 = __builtin_amdgcn_wmma_f32_16x16x32_bf16(false, a_h, false, b0h,
                                                           (short)0, acc0, false, false);
            acc0 = __builtin_amdgcn_wmma_f32_16x16x32_bf16(false, a_h, false, b0l,
                                                           (short)0, acc0, false, false);
            acc0 = __builtin_amdgcn_wmma_f32_16x16x32_bf16(false, a_l, false, b0h,
                                                           (short)0, acc0, false, false);

            const v16bf b1h = *(const v16bf*)(sh_hi + base1);
            const v16bf b1l = *(const v16bf*)(sh_lo + base1);
            acc1 = __builtin_amdgcn_wmma_f32_16x16x32_bf16(false, a_h, false, b1h,
                                                           (short)0, acc1, false, false);
            acc1 = __builtin_amdgcn_wmma_f32_16x16x32_bf16(false, a_h, false, b1l,
                                                           (short)0, acc1, false, false);
            acc1 = __builtin_amdgcn_wmma_f32_16x16x32_bf16(false, a_l, false, b1h,
                                                           (short)0, acc1, false, false);
        }
    }

    // C/D layout: VGPR r -> o = otile*16 + r + 8*(lane>=16); column n = lane%16
    const int n0 = (pxh << 5) + n16;
    const int obase = otile * 16 + ((lane >> 4) << 3);
    float* op = out + (size_t)b * On * Hn * Wn;
#pragma unroll
    for (int r = 0; r < 8; ++r) {
        const int o = obase + r;
        const float bias = agg_b[b * On + o];
        const size_t p = ((size_t)o << 16) + ((size_t)ty << 8) + x0 + n0;
        op[p]      = acc0[r] + bias;
        op[p + 16] = acc1[r] + bias;
    }
}

extern "C" void kernel_launch(void* const* d_in, const int* in_sizes, int n_in,
                              void* d_out, int out_size, void* d_ws, size_t ws_size,
                              hipStream_t stream) {
    const float* x      = (const float*)d_in[0];
    const float* fc1_w  = (const float*)d_in[1];
    const float* fc2_w  = (const float*)d_in[2];
    const float* fc2_b  = (const float*)d_in[3];
    const float* weight = (const float*)d_in[4];
    const float* bias_k = (const float*)d_in[5];
    float* out = (float*)d_out;

    char* ws = (char*)d_ws;
    float* pooled = (float*)(ws + 0);                    //  4096 B
    float* att    = (float*)(ws + 4096);                 //   256 B
    float* agg_b  = (float*)(ws + 4352);                 //  4096 B
    unsigned short* Ahi = (unsigned short*)(ws + 8448);  // 1179648 B (32B-aligned)
    unsigned short* Alo = (unsigned short*)(ws + 8448 + 1179648);

    pool_kernel<<<dim3(Bn * Cn), dim3(256), 0, stream>>>(x, pooled);
    attn_kernel<<<dim3(1), dim3(32), 0, stream>>>(pooled, fc1_w, fc2_w, fc2_b,
                                                  bias_k, att, agg_b);
    afrag_kernel<<<dim3(589824 / 256), dim3(256), 0, stream>>>(weight, att, Ahi, Alo);
    conv_kernel<<<dim3((Wn / TW) * Hn, Bn), dim3(256), 0, stream>>>(x, Ahi, Alo,
                                                                    agg_b, out);
}
